// mHCBlock_80951543595213
// MI455X (gfx1250) — matrix operand order repaired and checked by reading, not compiled, not run on hardware
//
#include <hip/hip_runtime.h>
#include <hip/hip_bf16.h>
#include <math.h>

#define DEVI __device__ __forceinline__

typedef __attribute__((ext_vector_type(16))) __bf16 v16bf;
typedef __attribute__((ext_vector_type(8)))  __bf16 bf16x8;
typedef __attribute__((ext_vector_type(8)))  float  v8f;

static constexpr int Bsz   = 8192;
static constexpr int Ns    = 4;
static constexpr int Cd    = 768;
static constexpr int NC    = 3072;       // Ns*Cd
static constexpr int HID   = 3072;       // 4*Cd
static constexpr int Mrows = Bsz * Ns;   // 32768
static constexpr float EPS_RMS = 1.1920929e-7f;
static constexpr float EPS_SK  = 1e-6f;
static constexpr int   SK_ITERS = 20;

DEVI unsigned short f32_to_bf16(float f) {
  unsigned int u = __float_as_uint(f);
  unsigned int r = (u + 0x7FFFu + ((u >> 16) & 1u)) >> 16;  // RNE
  return (unsigned short)r;
}

// Build a v16bf WMMA fragment from two 16-byte LDS chunks.
DEVI v16bf frag_from(const unsigned short* p0, const unsigned short* p1) {
  bf16x8 lo = *(const bf16x8*)p0;
  bf16x8 hi = *(const bf16x8*)p1;
  v16bf r;
#pragma unroll
  for (int i = 0; i < 8; ++i) { r[i] = lo[i]; r[i + 8] = hi[i]; }
  return r;
}

// Generic (shared) pointer -> 32-bit LDS byte address for async-to-LDS ops.
typedef __attribute__((address_space(3))) unsigned short lds_us;
DEVI unsigned lds_addr(unsigned short* p) {
  return (unsigned)(unsigned long long)(lds_us*)p;
}

// Issue one 64-byte stripe (4 x b128) of A and B for a K-stage.
// INST_OFFSET is added to both the LDS and the global address (ISA 10. async),
// so a single base covers the stripe on both sides.
DEVI void async_stage(unsigned ldsA, const unsigned short* gA,
                      unsigned ldsB, const unsigned short* gB) {
  asm volatile(
      "global_load_async_to_lds_b128 %0, %1, off\n\t"
      "global_load_async_to_lds_b128 %0, %1, off offset:16\n\t"
      "global_load_async_to_lds_b128 %0, %1, off offset:32\n\t"
      "global_load_async_to_lds_b128 %0, %1, off offset:48\n\t"
      "global_load_async_to_lds_b128 %2, %3, off\n\t"
      "global_load_async_to_lds_b128 %2, %3, off offset:16\n\t"
      "global_load_async_to_lds_b128 %2, %3, off offset:32\n\t"
      "global_load_async_to_lds_b128 %2, %3, off offset:48"
      :: "v"(ldsA), "v"(gA), "v"(ldsB), "v"(gB) : "memory");
}

// ---------------------------------------------------------------------------
// Kernel 1: fp32 -> bf16 weight conversion (W1 [3072,768], W2 [768,3072])
// ---------------------------------------------------------------------------
__global__ __launch_bounds__(256)
void cvt_weights_kernel(const float* __restrict__ w1, const float* __restrict__ w2,
                        unsigned short* __restrict__ W1b, unsigned short* __restrict__ W2b,
                        int n_each) {
  int i = blockIdx.x * 256 + threadIdx.x;
  if (i < n_each) {
    W1b[i] = f32_to_bf16(w1[i]);
  } else {
    int j = i - n_each;
    if (j < n_each) W2b[j] = f32_to_bf16(w2[j]);
  }
}

// ---------------------------------------------------------------------------
// Kernel 2: per-row fused  RMSNorm -> gate logits -> sigmoid / Sinkhorn ->
//           residual mix (written straight into d_out) + bf16 GEMM-A matrix.
// One 256-thread block (8 waves) per batch row.
// ---------------------------------------------------------------------------
__global__ __launch_bounds__(256)
void gates_mix_kernel(const float* __restrict__ x,
                      const float* __restrict__ norm_w,
                      const float* __restrict__ w_pre,  const float* __restrict__ b_pre,
                      const float* __restrict__ w_post, const float* __restrict__ b_post,
                      const float* __restrict__ w_res,  const float* __restrict__ b_res,
                      const float* __restrict__ alpha_pre,
                      const float* __restrict__ alpha_post,
                      const float* __restrict__ alpha_res,
                      float* __restrict__ out_res,          // d_out [B,4,768]
                      unsigned short* __restrict__ A_bf,    // [32768,768] bf16
                      float* __restrict__ Hpost_g)          // [32768]
{
  const int b   = blockIdx.x;
  const int tid = threadIdx.x;
  const float* xr = x + (size_t)b * NC;

  __shared__ float red[8 * 25];
  __shared__ float tot[25];
  __shared__ float sHres[16];
  __shared__ float sHpre[4];

  // one pass: sumsq + 24 gate dot-product partials
  float acc[25];
#pragma unroll
  for (int o = 0; o < 25; ++o) acc[o] = 0.f;

  for (int idx = tid; idx < NC; idx += 256) {
    float xv = xr[idx];
    float xs = xv * norm_w[idx];
    acc[0] += xv * xv;
#pragma unroll
    for (int o = 0; o < 4; ++o)  acc[1 + o] += xs * w_pre [o * NC + idx];
#pragma unroll
    for (int o = 0; o < 4; ++o)  acc[5 + o] += xs * w_post[o * NC + idx];
#pragma unroll
    for (int o = 0; o < 16; ++o) acc[9 + o] += xs * w_res [o * NC + idx];
  }

  // wave32 reduction, then cross-wave via LDS
#pragma unroll
  for (int o = 0; o < 25; ++o) {
    float v = acc[o];
#pragma unroll
    for (int s = 16; s > 0; s >>= 1) v += __shfl_xor(v, s, 32);
    acc[o] = v;
  }
  const int lane = tid & 31, wave = tid >> 5;
  if (lane == 0)
    for (int o = 0; o < 25; ++o) red[wave * 25 + o] = acc[o];
  __syncthreads();
  if (tid < 25) {
    float s = 0.f;
    for (int w = 0; w < 8; ++w) s += red[w * 25 + tid];
    tot[tid] = s;
  }
  __syncthreads();

  if (tid == 0) {
    float rs  = rsqrtf(tot[0] / (float)NC + EPS_RMS);
    float ap  = alpha_pre[0], apo = alpha_post[0], ar = alpha_res[0];
#pragma unroll
    for (int i = 0; i < 4; ++i) {
      float pre  = ap  * rs * tot[1 + i] + b_pre[i];
      sHpre[i]   = 1.f / (1.f + expf(-pre));
      float post = apo * rs * tot[5 + i] + b_post[i];
      Hpost_g[b * 4 + i] = 2.f / (1.f + expf(-post));
    }
    float M[16];
#pragma unroll
    for (int k = 0; k < 16; ++k)
      M[k] = expf(ar * rs * tot[9 + k] + b_res[k]);
    for (int it = 0; it < SK_ITERS; ++it) {
#pragma unroll
      for (int i = 0; i < 4; ++i) {
        float inv = 1.f / (M[i*4] + M[i*4+1] + M[i*4+2] + M[i*4+3] + EPS_SK);
        M[i*4] *= inv; M[i*4+1] *= inv; M[i*4+2] *= inv; M[i*4+3] *= inv;
      }
#pragma unroll
      for (int j = 0; j < 4; ++j) {
        float inv = 1.f / (M[j] + M[4+j] + M[8+j] + M[12+j] + EPS_SK);
        M[j] *= inv; M[4+j] *= inv; M[8+j] *= inv; M[12+j] *= inv;
      }
    }
#pragma unroll
    for (int k = 0; k < 16; ++k) sHres[k] = M[k];
  }
  __syncthreads();

  // residual mix (fp32, into d_out) + pre-gated mix (bf16, GEMM-A)
  for (int c = tid; c < Cd; c += 256) {
    float x0 = xr[0*Cd + c], x1 = xr[1*Cd + c], x2 = xr[2*Cd + c], x3 = xr[3*Cd + c];
    float g0 = x0 * sHpre[0], g1 = x1 * sHpre[1], g2 = x2 * sHpre[2], g3 = x3 * sHpre[3];
#pragma unroll
    for (int i = 0; i < 4; ++i) {
      float h0 = sHres[i*4+0], h1 = sHres[i*4+1], h2 = sHres[i*4+2], h3 = sHres[i*4+3];
      float r = h0*x0 + h1*x1 + h2*x2 + h3*x3;
      float s = h0*g0 + h1*g1 + h2*g2 + h3*g3;
      size_t off = (size_t)(b * 4 + i) * Cd + c;
      out_res[off] = r;
      A_bf[off]    = f32_to_bf16(s);
    }
  }
}

// ---------------------------------------------------------------------------
// Kernel 3/4: async double-buffered bf16 WMMA GEMM  D = A[M,K] * Bw[N,K]^T
//   GELU_STORE_BF16 = true : out_bf[m*N+n] = bf16(gelu(acc + bias[n]))
//   GELU_STORE_BF16 = false: out_f32[m*N+n] += row_scale[m]*(acc + bias[n])
// 256 threads = 8 waves (2 x 4); block tile 128x128; wave tile 64x32;
// K-stage 64 (two WMMA k-steps), LDS double-buffered, fills via
// GLOBAL_LOAD_ASYNC_TO_LDS_B128 tracked by ASYNCcnt.
// ---------------------------------------------------------------------------
template <bool GELU_STORE_BF16>
__global__ __launch_bounds__(256)
void gemm_bf16_kernel(const unsigned short* __restrict__ A,   // [M,K] bf16
                      const unsigned short* __restrict__ Bw,  // [N,K] bf16
                      const float* __restrict__ bias,         // [N]
                      const float* __restrict__ row_scale,    // [M] (epilogue 2)
                      unsigned short* __restrict__ out_bf,    // [M,N]
                      float* __restrict__ out_f32,            // [M,N]
                      int M, int N, int K)
{
  constexpr int SA = 72;  // padded LDS row stride (64 K-elems + 8 pad)
  __shared__ __align__(16) unsigned short As[2][128 * SA];
  __shared__ __align__(16) unsigned short Bs[2][128 * SA];

  const int tid  = threadIdx.x;
  const int lane = tid & 31;
  const int wave = tid >> 5;
  const int wm = wave & 1;        // 2 waves along M
  const int wn = wave >> 1;       // 4 waves along N
  const int lr = lane & 15;       // A row / B col / D col within 16-tile
  const int kh = lane >> 4;       // K-half selector

  const int m0 = blockIdx.x * 128;
  const int n0 = blockIdx.y * 128;

  v8f acc[4][2];
#pragma unroll
  for (int a = 0; a < 4; ++a)
#pragma unroll
    for (int q = 0; q < 2; ++q)
#pragma unroll
      for (int v = 0; v < 8; ++v) acc[a][q][v] = 0.f;

  // async staging map: each thread owns a 64B stripe (half a row's K-stage)
  const int sr = tid >> 1;            // 0..127 : tile row
  const int sc = (tid & 1) * 32;      // element col base within stage (0|32)
  const unsigned short* gA = A  + (size_t)(m0 + sr) * K + sc;
  const unsigned short* gB = Bw + (size_t)(n0 + sr) * K + sc;
  unsigned ldsA[2], ldsB[2];
#pragma unroll
  for (int bb = 0; bb < 2; ++bb) {
    ldsA[bb] = lds_addr(&As[bb][sr * SA + sc]);
    ldsB[bb] = lds_addr(&Bs[bb][sr * SA + sc]);
  }

  const int ktiles = K >> 6;          // K-stages of 64 (K is 768 or 3072)

  // prologue: stages 0 and 1 in flight
  async_stage(ldsA[0], gA,      ldsB[0], gB);
  async_stage(ldsA[1], gA + 64, ldsB[1], gB + 64);

  for (int kt = 0; kt < ktiles; ++kt) {
    // retire stage kt (async loads complete in order; newer stage may remain)
    if (kt + 1 < ktiles) asm volatile("s_wait_asynccnt 8" ::: "memory");
    else                 asm volatile("s_wait_asynccnt 0" ::: "memory");
    __syncthreads();

    unsigned short* curA = As[kt & 1];
    unsigned short* curB = Bs[kt & 1];
#pragma unroll
    for (int s = 0; s < 2; ++s) {     // two 16x16x32 k-steps per stage
      v16bf af[4], bfr[2];
#pragma unroll
      for (int t = 0; t < 4; ++t) {
        unsigned short* p = &curA[(wm * 64 + t * 16 + lr) * SA + s * 32 + kh * 8];
        af[t] = frag_from(p, p + 16);
      }
#pragma unroll
      for (int t = 0; t < 2; ++t) {
        unsigned short* p = &curB[(wn * 32 + t * 16 + lr) * SA + s * 32 + kh * 8];
        bfr[t] = frag_from(p, p + 16);
      }
#pragma unroll
      for (int a = 0; a < 4; ++a)
#pragma unroll
        for (int q = 0; q < 2; ++q)
          acc[a][q] = __builtin_amdgcn_wmma_f32_16x16x32_bf16(
              false, af[a], false, bfr[q], (short)0, acc[a][q], false, false);
    }
    __syncthreads();                  // everyone done reading buffer kt&1

    if (kt + 2 < ktiles)              // refill the buffer we just drained
      async_stage(ldsA[kt & 1], gA + (size_t)(kt + 2) * 64,
                  ldsB[kt & 1], gB + (size_t)(kt + 2) * 64);
  }

  // epilogue — D layout: n = lane&15 ; m = v + 8*(lane>>4)
  const int mW = m0 + wm * 64;
  const int nW = n0 + wn * 32;
  float biasv[2];
#pragma unroll
  for (int q = 0; q < 2; ++q) biasv[q] = bias[nW + q * 16 + lr];

#pragma unroll
  for (int a = 0; a < 4; ++a) {
#pragma unroll
    for (int q = 0; q < 2; ++q) {
#pragma unroll
      for (int v = 0; v < 8; ++v) {
        int m = mW + a * 16 + v + 8 * kh;
        int n = nW + q * 16 + lr;
        float val = acc[a][q][v] + biasv[q];
        size_t off = (size_t)m * N + n;
        if (GELU_STORE_BF16) {
          float g = 0.5f * val * (1.f + erff(val * 0.70710678118654752f));
          out_bf[off] = f32_to_bf16(g);
        } else {
          out_f32[off] += val * row_scale[m];   // residual already in d_out
        }
      }
    }
  }
}

// ---------------------------------------------------------------------------
extern "C" void kernel_launch(void* const* d_in, const int* in_sizes, int n_in,
                              void* d_out, int out_size, void* d_ws, size_t ws_size,
                              hipStream_t stream) {
  const float* x         = (const float*)d_in[0];
  const float* norm_w    = (const float*)d_in[1];
  const float* w_pre     = (const float*)d_in[2];
  const float* b_pre     = (const float*)d_in[3];
  const float* w_post    = (const float*)d_in[4];
  const float* b_post    = (const float*)d_in[5];
  const float* w_res     = (const float*)d_in[6];
  const float* b_res     = (const float*)d_in[7];
  const float* alpha_pre = (const float*)d_in[8];
  const float* alpha_post= (const float*)d_in[9];
  const float* alpha_res = (const float*)d_in[10];
  const float* w1        = (const float*)d_in[11];
  const float* b1        = (const float*)d_in[12];
  const float* w2        = (const float*)d_in[13];
  const float* b2        = (const float*)d_in[14];
  float* out = (float*)d_out;

  // workspace carve-up
  char* ws = (char*)d_ws;
  size_t off = 0;
  auto alloc = [&](size_t bytes) {
    char* p = ws + off;
    off += (bytes + 255) & ~(size_t)255;
    return p;
  };
  unsigned short* Abf   = (unsigned short*)alloc((size_t)Mrows * Cd  * 2);  //  50 MB
  unsigned short* Hbf   = (unsigned short*)alloc((size_t)Mrows * HID * 2);  // 201 MB
  unsigned short* W1b   = (unsigned short*)alloc((size_t)HID * Cd * 2);
  unsigned short* W2b   = (unsigned short*)alloc((size_t)Cd * HID * 2);
  float*          Hpost = (float*)alloc((size_t)Mrows * sizeof(float));

  const int n_each = HID * Cd;  // 2359296
  hipLaunchKernelGGL(cvt_weights_kernel, dim3((2 * n_each) / 256), dim3(256), 0, stream,
                     w1, w2, W1b, W2b, n_each);

  hipLaunchKernelGGL(gates_mix_kernel, dim3(Bsz), dim3(256), 0, stream,
                     x, norm_w, w_pre, b_pre, w_post, b_post, w_res, b_res,
                     alpha_pre, alpha_post, alpha_res, out, Abf, Hpost);

  // h = gelu(A @ W1^T + b1)  -> bf16
  hipLaunchKernelGGL((gemm_bf16_kernel<true>), dim3(Mrows / 128, HID / 128), dim3(256), 0, stream,
                     Abf, W1b, b1, nullptr, Hbf, nullptr, Mrows, HID, Cd);

  // d_out += H_post * (h @ W2^T + b2)
  hipLaunchKernelGGL((gemm_bf16_kernel<false>), dim3(Mrows / 128, Cd / 128), dim3(256), 0, stream,
                     Hbf, W2b, b2, Hpost, nullptr, out, Mrows, Cd, HID);
}
